// LayerNormBasicLSTMCell_24927990186081
// MI455X (gfx1250) — matrix operand order repaired and checked by reading, not compiled, not run on hardware
//
#include <hip/hip_runtime.h>
#include <hip/hip_bf16.h>

// Problem sizes (fixed by the reference)
#define B_SZ 8192
#define H_SZ 1024
#define K_SZ 2048   // I + H
#define N_SZ 4096   // 4 * H

typedef __attribute__((ext_vector_type(16))) __bf16 v16bf;
typedef __attribute__((ext_vector_type(8)))  float  v8f;
typedef int v4i_vs __attribute__((vector_size(16)));   // matches builtin param

struct B32x2 { uint4 lo, hi; };

__device__ __forceinline__ v16bf make_frag(uint4 lo, uint4 hi) {
  B32x2 tmp{lo, hi};
  return __builtin_bit_cast(v16bf, tmp);
}

// float -> bf16 with round-to-nearest-even
__device__ __forceinline__ unsigned short f2bf(float f) {
  unsigned int u = __float_as_uint(f);
  u += 0x7FFFu + ((u >> 16) & 1u);
  return (unsigned short)(u >> 16);
}

// ---- gfx1250 async global->LDS path (guarded; falls back to reg staging) ----
#if __has_builtin(__builtin_amdgcn_global_load_async_to_lds_b128)
#define HAVE_ASYNC_LDS 1
__device__ __forceinline__ void async_cp16(const unsigned short* gsrc,
                                           unsigned short* ldst) {
  __builtin_amdgcn_global_load_async_to_lds_b128(
      (__attribute__((address_space(1))) v4i_vs*)gsrc,
      (__attribute__((address_space(3))) v4i_vs*)ldst,
      0, 0);
}
#endif

#if __has_builtin(__builtin_amdgcn_s_wait_asynccnt)
#define WAIT_ASYNC() __builtin_amdgcn_s_wait_asynccnt(0)
#else
#define WAIT_ASYNC() asm volatile("s_wait_asynccnt 0x0" ::: "memory")
#endif

// ---------------------------------------------------------------------------
// Kernel 1a: pack concat([inputs, h]) -> bf16 A[B, K]
// ---------------------------------------------------------------------------
__global__ void pack_concat_kernel(const float* __restrict__ x,
                                   const float* __restrict__ h,
                                   unsigned short* __restrict__ A) {
  size_t idx = (size_t)blockIdx.x * blockDim.x + threadIdx.x;   // < B*K
  size_t row = idx >> 11;            // / 2048
  int    col = (int)(idx & 2047);
  float v = (col < H_SZ) ? x[row * H_SZ + col] : h[row * H_SZ + (col - H_SZ)];
  A[idx] = f2bf(v);
}

// ---------------------------------------------------------------------------
// Kernel 1b: pack W[N, K] -> bf16 (same row-major layout, K contiguous)
// ---------------------------------------------------------------------------
__global__ void pack_w_kernel(const float* __restrict__ W,
                              unsigned short* __restrict__ Wb) {
  size_t idx = (size_t)blockIdx.x * blockDim.x + threadIdx.x;   // < N*K
  Wb[idx] = f2bf(W[idx]);
}

// ---------------------------------------------------------------------------
// Kernel 2: C[M,N] = A[M,K] * Wb[N,K]^T  via v_wmma_f32_16x16x32_bf16
// Block tile 128x128, BK=32, 256 threads (8 wave32), wave tile 32x64 (2x4 WMMA)
// Double-buffered LDS; async global->LDS when available.
// ---------------------------------------------------------------------------
__global__ void __launch_bounds__(256)
gemm_bf16_wmma_kernel(const unsigned short* __restrict__ A,
                      const unsigned short* __restrict__ Bm,
                      float* __restrict__ C) {
  constexpr int LDSS = 40;   // 32 + 8 pad elements (80B row stride, 16B aligned)
  __shared__ __align__(16) unsigned short As[2][128 * LDSS];
  __shared__ __align__(16) unsigned short Bs[2][128 * LDSS];   // 40 KiB total

  const int t        = threadIdx.x;
  const int wave     = t >> 5;
  const int lane     = t & 31;
  const int lane15   = lane & 15;
  const int laneHalf = lane >> 4;          // 0 or 1
  const int wm       = wave >> 1;          // 0..3 -> M offset wm*32
  const int wn       = wave & 1;           // 0..1 -> N offset wn*64

  const int m_blk = blockIdx.y * 128;
  const int n_blk = blockIdx.x * 128;

  // global->LDS: each thread moves one 32B half-row (16 bf16) per matrix
  const int ldRow  = t >> 1;               // 0..127
  const int ldOff  = (t & 1) * 16;         // element offset 0 or 16

  const unsigned short* aSrc = A  + (size_t)(m_blk + ldRow) * K_SZ + ldOff;
  const unsigned short* bSrc = Bm + (size_t)(n_blk + ldRow) * K_SZ + ldOff;
  const int ldsOff = ldRow * LDSS + ldOff;

  v8f acc[2][4] = {};

  constexpr int NSTEP = K_SZ / 32;

#ifdef HAVE_ASYNC_LDS
  // ---- async double-buffered pipeline ----
  async_cp16(aSrc,     &As[0][ldsOff]);
  async_cp16(aSrc + 8, &As[0][ldsOff + 8]);
  async_cp16(bSrc,     &Bs[0][ldsOff]);
  async_cp16(bSrc + 8, &Bs[0][ldsOff + 8]);
  WAIT_ASYNC();
  __syncthreads();
#else
  // ---- register-staged double-buffered pipeline ----
  uint4 ra0 = *(const uint4*)(aSrc);
  uint4 ra1 = *(const uint4*)(aSrc + 8);
  uint4 rb0 = *(const uint4*)(bSrc);
  uint4 rb1 = *(const uint4*)(bSrc + 8);
  *(uint4*)&As[0][ldsOff]     = ra0;
  *(uint4*)&As[0][ldsOff + 8] = ra1;
  *(uint4*)&Bs[0][ldsOff]     = rb0;
  *(uint4*)&Bs[0][ldsOff + 8] = rb1;
  __syncthreads();
#endif

  for (int step = 0; step < NSTEP; ++step) {
    const int cur  = step & 1;
    const int nxt  = cur ^ 1;
    const bool more = (step + 1) < NSTEP;
    const int koff  = (step + 1) * 32;

    // issue next tile's fetch so it overlaps with this tile's WMMAs
#ifdef HAVE_ASYNC_LDS
    if (more) {
      async_cp16(aSrc + koff,     &As[nxt][ldsOff]);
      async_cp16(aSrc + koff + 8, &As[nxt][ldsOff + 8]);
      async_cp16(bSrc + koff,     &Bs[nxt][ldsOff]);
      async_cp16(bSrc + koff + 8, &Bs[nxt][ldsOff + 8]);
    }
#else
    uint4 na0, na1, nb0, nb1;
    if (more) {
      na0 = *(const uint4*)(aSrc + koff);
      na1 = *(const uint4*)(aSrc + koff + 8);
      nb0 = *(const uint4*)(bSrc + koff);
      nb1 = *(const uint4*)(bSrc + koff + 8);
    }
#endif

    // ---- compute on current buffer ----
    const unsigned short* as = As[cur];
    const unsigned short* bs = Bs[cur];
    v16bf af[2], bf[4];
#pragma unroll
    for (int ti = 0; ti < 2; ++ti) {
      // A operand 16x32: lane<16 -> K[0:8)+K[16:24); lane>=16 -> K[8:16)+K[24:32)
      int base = (wm * 32 + ti * 16 + lane15) * LDSS + laneHalf * 8;
      af[ti] = make_frag(*(const uint4*)&as[base], *(const uint4*)&as[base + 16]);
    }
#pragma unroll
    for (int tj = 0; tj < 4; ++tj) {
      // B operand 32x16: lane<16 -> K[0:8)+K[16:24); lane>=16 -> K[8:16)+K[24:32)
      int base = (wn * 64 + tj * 16 + lane15) * LDSS + laneHalf * 8;
      bf[tj] = make_frag(*(const uint4*)&bs[base], *(const uint4*)&bs[base + 16]);
    }
#pragma unroll
    for (int ti = 0; ti < 2; ++ti)
#pragma unroll
      for (int tj = 0; tj < 4; ++tj)
        acc[ti][tj] = __builtin_amdgcn_wmma_f32_16x16x32_bf16(
            false, af[ti], false, bf[tj], (short)0, acc[ti][tj], false, false);

    // ---- publish next buffer ----
#ifdef HAVE_ASYNC_LDS
    if (more) WAIT_ASYNC();
#else
    if (more) {
      *(uint4*)&As[nxt][ldsOff]     = na0;
      *(uint4*)&As[nxt][ldsOff + 8] = na1;
      *(uint4*)&Bs[nxt][ldsOff]     = nb0;
      *(uint4*)&Bs[nxt][ldsOff + 8] = nb1;
    }
#endif
    __syncthreads();
  }

  // C/D layout: M = (lane>>4)*8 + r, N = lane&15
#pragma unroll
  for (int ti = 0; ti < 2; ++ti)
#pragma unroll
    for (int tj = 0; tj < 4; ++tj) {
      int m0 = m_blk + wm * 32 + ti * 16 + laneHalf * 8;
      int n0 = n_blk + wn * 64 + tj * 16 + lane15;
      float* dst = C + (size_t)m0 * N_SZ + n0;
#pragma unroll
      for (int r = 0; r < 8; ++r)
        dst[(size_t)r * N_SZ] = acc[ti][tj][r];
    }
}

// ---------------------------------------------------------------------------
// Kernel 3: per-row LayerNorms + gates + cell update. One block per batch row.
// ---------------------------------------------------------------------------
__device__ __forceinline__ void block_reduce2(float& s, float& s2, float* red,
                                              int t) {
#pragma unroll
  for (int off = 16; off > 0; off >>= 1) {
    s  += __shfl_xor(s, off, 32);
    s2 += __shfl_xor(s2, off, 32);
  }
  int wave = t >> 5;
  if ((t & 31) == 0) { red[wave * 2] = s; red[wave * 2 + 1] = s2; }
  __syncthreads();
  float S = 0.f, S2 = 0.f;
#pragma unroll
  for (int w = 0; w < 8; ++w) { S += red[w * 2]; S2 += red[w * 2 + 1]; }
  __syncthreads();
  s = S; s2 = S2;
}

__global__ void __launch_bounds__(256)
lstm_epilogue_kernel(const float* __restrict__ pre,
                     const float* __restrict__ c,
                     const float* __restrict__ gamma4,
                     const float* __restrict__ beta4,
                     const float* __restrict__ cg,
                     const float* __restrict__ cb,
                     float* __restrict__ out_h,
                     float* __restrict__ out_c) {
  const int b = blockIdx.x;
  const int t = threadIdx.x;         // 256 threads, 4 elems each per gate
  __shared__ float red[16];

  const float* row = pre + (size_t)b * N_SZ;

  float v[4][4];
#pragma unroll
  for (int g = 0; g < 4; ++g) {
    float s = 0.f, s2 = 0.f;
#pragma unroll
    for (int i = 0; i < 4; ++i) {
      float x = row[g * H_SZ + i * 256 + t];
      v[g][i] = x;
      s += x; s2 += x * x;
    }
    block_reduce2(s, s2, red, t);
    float mu  = s * (1.f / (float)H_SZ);
    float var = s2 * (1.f / (float)H_SZ) - mu * mu;
    float rs  = rsqrtf(var + 1e-5f);
#pragma unroll
    for (int i = 0; i < 4; ++i) {
      int hi = i * 256 + t;
      v[g][i] = (v[g][i] - mu) * rs * gamma4[g * H_SZ + hi] + beta4[g * H_SZ + hi];
    }
  }

  // gates: i=sigmoid, j=tanh, f=sigmoid, o=sigmoid; cand = i*j + f*c
  float cand[4];
  float s = 0.f, s2 = 0.f;
#pragma unroll
  for (int i = 0; i < 4; ++i) {
    int hi = i * 256 + t;
    float ig = 1.f / (1.f + __expf(-v[0][i]));
    float jg = tanhf(v[1][i]);
    float fg = 1.f / (1.f + __expf(-v[2][i]));
    float cc = ig * jg + fg * c[(size_t)b * H_SZ + hi];
    cand[i] = cc;
    s += cc; s2 += cc * cc;
    v[3][i] = 1.f / (1.f + __expf(-v[3][i]));   // o gate
  }
  block_reduce2(s, s2, red, t);
  float mu  = s * (1.f / (float)H_SZ);
  float var = s2 * (1.f / (float)H_SZ) - mu * mu;
  float rs  = rsqrtf(var + 1e-5f);
#pragma unroll
  for (int i = 0; i < 4; ++i) {
    int hi = i * 256 + t;
    float nc = (cand[i] - mu) * rs * cg[hi] + cb[hi];
    out_c[(size_t)b * H_SZ + hi] = nc;
    out_h[(size_t)b * H_SZ + hi] = v[3][i] * tanhf(nc);
  }
}

// ---------------------------------------------------------------------------
extern "C" void kernel_launch(void* const* d_in, const int* in_sizes, int n_in,
                              void* d_out, int out_size, void* d_ws, size_t ws_size,
                              hipStream_t stream) {
  const float* inputs = (const float*)d_in[0];
  const float* h      = (const float*)d_in[1];
  const float* c      = (const float*)d_in[2];
  const float* W      = (const float*)d_in[3];
  const float* gamma4 = (const float*)d_in[4];
  const float* beta4  = (const float*)d_in[5];
  const float* cg     = (const float*)d_in[6];
  const float* cb     = (const float*)d_in[7];

  char* ws = (char*)d_ws;
  unsigned short* Abf = (unsigned short*)ws;                        // 32 MiB
  unsigned short* Wbf = (unsigned short*)(ws + ((size_t)32 << 20)); // 16 MiB
  float*          pre = (float*)(ws + ((size_t)48 << 20));          // 128 MiB

  float* out_h = (float*)d_out;
  float* out_c = out_h + (size_t)B_SZ * H_SZ;

  pack_concat_kernel<<<(B_SZ * (size_t)K_SZ) / 256, 256, 0, stream>>>(inputs, h, Abf);
  pack_w_kernel<<<((size_t)N_SZ * K_SZ) / 256, 256, 0, stream>>>(W, Wbf);
  gemm_bf16_wmma_kernel<<<dim3(N_SZ / 128, B_SZ / 128), 256, 0, stream>>>(Abf, Wbf, pre);
  lstm_epilogue_kernel<<<B_SZ, 256, 0, stream>>>(pre, c, gamma4, beta4, cg, cb,
                                                 out_h, out_c);
}
